// GraphGRUSeq2Vec_12309376270431
// MI455X (gfx1250) — matrix operand order repaired and checked by reading, not compiled, not run on hardware
//
#include <hip/hip_runtime.h>
#include <cstdint>
#include <cstddef>

// ---------------------------------------------------------------------------
// GraphGRU seq2vec for MI455X (gfx1250, wave32, WMMA).
// Heavy ops (A@x gconv, D->3D gate projections) run through
// v_wmma_f32_16x16x32_bf16; accumulation + GRU nonlinearity in f32.
// ---------------------------------------------------------------------------

typedef __bf16 bf16;
typedef __attribute__((ext_vector_type(16))) __bf16          v16bf;
typedef __attribute__((ext_vector_type(8)))  __bf16          v8bf;
typedef __attribute__((ext_vector_type(8)))  float           v8f;
typedef __attribute__((ext_vector_type(8)))  unsigned short  v8u16;

#define Bb 32
#define Tt 12
#define Nn 2048
#define Dd 64
#define Ll 2
#define Hh 12
#define BN (Bb * Nn)      // 65536 rows per "batch panel"
#define G2 (2 * Bb)       // xn-half + h-half stacked

// ---------------------------------------------------------------------------
// one-shot converts
// ---------------------------------------------------------------------------
__global__ void k_f2bf(const float* __restrict__ in, bf16* __restrict__ out, int n) {
  int i = blockIdx.x * blockDim.x + threadIdx.x;
  if (i < n) out[i] = (bf16)in[i];
}

// Wi/Wh [L, D, 3D] row-major  ->  bf16 [L, 3D, D] (column-major per matrix),
// so WMMA B-fragments become two contiguous 16B loads per lane.
__global__ void k_convW(const float* __restrict__ in, bf16* __restrict__ out) {
  int i = blockIdx.x * blockDim.x + threadIdx.x;
  if (i >= Ll * Dd * 3 * Dd) return;
  int l = i / (Dd * 3 * Dd);
  int r = i % (Dd * 3 * Dd);
  int k = r / (3 * Dd);
  int c = r % (3 * Dd);
  out[(size_t)l * (Dd * 3 * Dd) + (size_t)c * Dd + k] = (bf16)in[i];
}

__global__ void k_zero(float* __restrict__ p, int n) {
  int i = blockIdx.x * blockDim.x + threadIdx.x;
  if (i < n) p[i] = 0.f;
}

// ---------------------------------------------------------------------------
// LayerNorm (+ fused embedding for layer 0). One 64-thread block per row.
// Writes xn (bf16) into Xcat[0:BN] and bf16(h[l]) into Xcat[BN:2BN].
// ---------------------------------------------------------------------------
__global__ void k_ln(const float* __restrict__ data, const int* __restrict__ timei,
                     const int* __restrict__ wdayi, const float* __restrict__ W_data,
                     const float* __restrict__ b_data, const float* __restrict__ emb_time,
                     const float* __restrict__ emb_wd, const float* __restrict__ ln_g,
                     const float* __restrict__ ln_b, const float* __restrict__ h_in,
                     const float* __restrict__ h_self, bf16* __restrict__ Xcat,
                     int l, int t) {
  int row = blockIdx.x;          // 0..BN-1
  int d   = threadIdx.x;         // 0..63
  int b   = row >> 11;
  int n   = row & (Nn - 1);
  float x;
  if (l == 0) {
    int ti = timei[b * Tt + t];
    int wd = wdayi[b * Tt + t];
    x = data[(size_t)(b * Tt + t) * Nn + n] * W_data[d] + b_data[d]
        + emb_time[ti * Dd + d] + emb_wd[wd * Dd + d];
  } else {
    x = h_in[(size_t)row * Dd + d];
  }
  __shared__ float s1[64], s2[64];
  s1[d] = x; s2[d] = x * x;
  __syncthreads();
  for (int off = 32; off > 0; off >>= 1) {
    if (d < off) { s1[d] += s1[d + off]; s2[d] += s2[d + off]; }
    __syncthreads();
  }
  float mean = s1[0] * (1.f / Dd);
  float var  = s2[0] * (1.f / Dd) - mean * mean;
  float inv  = rsqrtf(var + 1e-5f);
  float xn   = (x - mean) * inv * ln_g[l * Dd + d] + ln_b[l * Dd + d];
  Xcat[(size_t)row * Dd + d]        = (bf16)xn;
  Xcat[(size_t)(BN + row) * Dd + d] = (bf16)h_self[(size_t)row * Dd + d];
}

// ---------------------------------------------------------------------------
// Graph conv:  Y[g] = A(bf16)[2048x2048] @ Xcat[g](bf16)[2048x64], g in 0..63.
// Block: 256 thr = 8 waves; tile 256 (M) x 64 (N); K staged in chunks of 64.
// Each wave owns 32 rows (2 A-fragments, 8 accumulators) so every B-fragment
// read from LDS feeds two WMMAs. X K-chunk staged transposed via dword-packed
// (k,k+1) bf16 pairs -> ds_store_b32; B-fragments are contiguous ds_load_b128.
// A-fragments load straight from row-major global (L2-resident, 8 MB).
// ---------------------------------------------------------------------------
__global__ __launch_bounds__(256)
void k_gconv(const bf16* __restrict__ A_bf, const bf16* __restrict__ Xcat,
             bf16* __restrict__ Y) {
  __shared__ bf16 Xs[Dd * 64];                 // Xs[d*64 + k]  (8 KB)
  const int tid  = threadIdx.x;
  const int lane = tid & 31;
  const int wave = tid >> 5;
  const int half = lane >> 4;                  // ISA half-wave selector
  const int l16  = lane & 15;
  const int m0   = blockIdx.x * 256 + wave * 32;
  const bf16* Xb = Xcat + (size_t)blockIdx.y * (Nn * Dd);

  // staging map: thread copies k-pair (2 rows) x 8 d's; 8 lanes span one
  // 128B row -> coalesced global_load_b128 pairs.
  const int skp = tid >> 3;                    // 0..31 -> k = 2*skp
  const int sd0 = (tid & 7) * 8;               // 0,8,...,56

  v8f acc[2][4];
  #pragma unroll
  for (int r = 0; r < 2; ++r)
    #pragma unroll
    for (int nt = 0; nt < 4; ++nt)
      #pragma unroll
      for (int j = 0; j < 8; ++j) acc[r][nt][j] = 0.f;

  const int arow0 = m0 + l16;
  for (int k0 = 0; k0 < Nn; k0 += 64) {
    __syncthreads();
    {   // stage [64 x 64] K-chunk transposed, dword-packed along k
      const bf16* src = Xb + (size_t)(k0 + 2 * skp) * Dd + sd0;
      v8u16 e = *(const v8u16*)(src);          // row k   (raw bf16 bits)
      v8u16 o = *(const v8u16*)(src + Dd);     // row k+1
      uint32_t* Xw = (uint32_t*)Xs;
      #pragma unroll
      for (int j = 0; j < 8; ++j)
        Xw[(sd0 + j) * 32 + skp] = (uint32_t)e[j] | ((uint32_t)o[j] << 16);
    }
    __syncthreads();
    if (k0 + 64 < Nn) {  // prefetch next X chunk + next A fragment rows
      __builtin_prefetch(Xb + (size_t)(k0 + 64 + 2 * skp) * Dd + sd0, 0, 1);
      __builtin_prefetch(A_bf + (size_t)arow0 * Nn + k0 + 64 + 8 * half, 0, 1);
      __builtin_prefetch(A_bf + (size_t)(arow0 + 16) * Nn + k0 + 64 + 8 * half, 0, 1);
    }
    #pragma unroll
    for (int ks = 0; ks < 2; ++ks) {
      const int kk = k0 + ks * 32;
      // A fragments (16x32 bf16): lanes 0-15 K=0..7,16..23; 16-31 K=8..15,24..31
      v16bf a[2];
      #pragma unroll
      for (int r = 0; r < 2; ++r) {
        const bf16* ap = A_bf + (size_t)(arow0 + 16 * r) * Nn + kk + 8 * half;
        v8bf lo = *(const v8bf*)(ap);
        v8bf hi = *(const v8bf*)(ap + 16);
        #pragma unroll
        for (int i = 0; i < 8; ++i) { a[r][i] = lo[i]; a[r][i + 8] = hi[i]; }
      }
      #pragma unroll
      for (int nt = 0; nt < 4; ++nt) {
        int col = nt * 16 + l16;
        const bf16* bp = &Xs[col * 64 + ks * 32 + 8 * half];
        v8bf blo = *(const v8bf*)(bp);
        v8bf bhi = *(const v8bf*)(bp + 16);
        v16bf bv;
        #pragma unroll
        for (int i = 0; i < 8; ++i) { bv[i] = blo[i]; bv[i + 8] = bhi[i]; }
        acc[0][nt] = __builtin_amdgcn_wmma_f32_16x16x32_bf16(
            false, a[0], false, bv, (short)0, acc[0][nt], false, false);
        acc[1][nt] = __builtin_amdgcn_wmma_f32_16x16x32_bf16(
            false, a[1], false, bv, (short)0, acc[1][nt], false, false);
      }
    }
  }
  // scatter C (8-VGPR layout: VGPR j -> M = j + 8*half, col = lane&15)
  bf16* Yb = Y + (size_t)blockIdx.y * (Nn * Dd);
  #pragma unroll
  for (int r = 0; r < 2; ++r)
    #pragma unroll
    for (int nt = 0; nt < 4; ++nt) {
      int col = nt * 16 + l16;
      #pragma unroll
      for (int j = 0; j < 8; ++j) {
        int rowc = m0 + 16 * r + j + 8 * half;
        Yb[(size_t)rowc * Dd + col] = (bf16)acc[r][nt][j];
      }
    }
}

// ---------------------------------------------------------------------------
// Gate kernel: gi = Y0 @ Wi + bi, gh = Y1 @ Wh + bh via WMMA, fused GRU update.
// r/z gates accumulate gi+gh into the same accumulator; i_n / h_n separate.
// One wave per 16-row tile; 48 WMMAs per wave.
// ---------------------------------------------------------------------------
__device__ __forceinline__ float sigf(float x) { return 1.f / (1.f + __expf(-x)); }

__device__ __forceinline__ v16bf load_bfrag(const bf16* __restrict__ W, int col,
                                            int ks, int half) {
  const bf16* p = W + (size_t)col * Dd + ks * 32 + 8 * half;
  v8bf lo = *(const v8bf*)(p);
  v8bf hi = *(const v8bf*)(p + 16);
  v16bf r;
  #pragma unroll
  for (int i = 0; i < 8; ++i) { r[i] = lo[i]; r[i + 8] = hi[i]; }
  return r;
}

__global__ __launch_bounds__(256)
void k_gate(const bf16* __restrict__ Y, const bf16* __restrict__ WiT,
            const bf16* __restrict__ WhT, const float* __restrict__ bi,
            const float* __restrict__ bh, float* __restrict__ hbuf, int l) {
  const int tid  = threadIdx.x;
  const int lane = tid & 31;
  const int wave = tid >> 5;
  const int half = lane >> 4;
  const int l16  = lane & 15;
  const int rowbase = (blockIdx.x * 8 + wave) * 16;

  const bf16*  Y0   = Y;                        // gconv(LN(inp))
  const bf16*  Y1   = Y + (size_t)BN * Dd;      // gconv(h[l])
  const bf16*  Wi_l = WiT + (size_t)l * Dd * 3 * Dd;
  const bf16*  Wh_l = WhT + (size_t)l * Dd * 3 * Dd;
  const float* bi_l = bi + l * 3 * Dd;
  const float* bh_l = bh + l * 3 * Dd;
  float*       h_l  = hbuf + (size_t)l * BN * Dd;

  v8f accR[4], accZ[4], accN[4], accHN[4];
  #pragma unroll
  for (int nt = 0; nt < 4; ++nt)
    #pragma unroll
    for (int j = 0; j < 8; ++j) {
      accR[nt][j] = 0.f; accZ[nt][j] = 0.f; accN[nt][j] = 0.f; accHN[nt][j] = 0.f;
    }

  const int arow = rowbase + l16;
  #pragma unroll
  for (int ks = 0; ks < 2; ++ks) {
    v16bf a0, a1;
    {
      v8bf lo = *(const v8bf*)(Y0 + (size_t)arow * Dd + ks * 32 + 8 * half);
      v8bf hi = *(const v8bf*)(Y0 + (size_t)arow * Dd + ks * 32 + 8 * half + 16);
      #pragma unroll
      for (int i = 0; i < 8; ++i) { a0[i] = lo[i]; a0[i + 8] = hi[i]; }
    }
    {
      v8bf lo = *(const v8bf*)(Y1 + (size_t)arow * Dd + ks * 32 + 8 * half);
      v8bf hi = *(const v8bf*)(Y1 + (size_t)arow * Dd + ks * 32 + 8 * half + 16);
      #pragma unroll
      for (int i = 0; i < 8; ++i) { a1[i] = lo[i]; a1[i + 8] = hi[i]; }
    }
    #pragma unroll
    for (int nt = 0; nt < 4; ++nt) {
      int col = nt * 16 + l16;
      v16bf b;
      b = load_bfrag(Wi_l, col, ks, half);            // r gate (input path)
      accR[nt] = __builtin_amdgcn_wmma_f32_16x16x32_bf16(false, a0, false, b, (short)0, accR[nt], false, false);
      b = load_bfrag(Wh_l, col, ks, half);            // r gate (hidden path)
      accR[nt] = __builtin_amdgcn_wmma_f32_16x16x32_bf16(false, a1, false, b, (short)0, accR[nt], false, false);
      b = load_bfrag(Wi_l, 64 + col, ks, half);       // z gate
      accZ[nt] = __builtin_amdgcn_wmma_f32_16x16x32_bf16(false, a0, false, b, (short)0, accZ[nt], false, false);
      b = load_bfrag(Wh_l, 64 + col, ks, half);
      accZ[nt] = __builtin_amdgcn_wmma_f32_16x16x32_bf16(false, a1, false, b, (short)0, accZ[nt], false, false);
      b = load_bfrag(Wi_l, 128 + col, ks, half);      // i_n
      accN[nt] = __builtin_amdgcn_wmma_f32_16x16x32_bf16(false, a0, false, b, (short)0, accN[nt], false, false);
      b = load_bfrag(Wh_l, 128 + col, ks, half);      // h_n
      accHN[nt] = __builtin_amdgcn_wmma_f32_16x16x32_bf16(false, a1, false, b, (short)0, accHN[nt], false, false);
    }
  }

  // fused GRU update (C layout: VGPR j -> row = rowbase + j + 8*half)
  #pragma unroll
  for (int nt = 0; nt < 4; ++nt) {
    int col = nt * 16 + l16;
    float bR  = bi_l[col] + bh_l[col];
    float bZ  = bi_l[64 + col] + bh_l[64 + col];
    float bIN = bi_l[128 + col];
    float bHN = bh_l[128 + col];
    #pragma unroll
    for (int j = 0; j < 8; ++j) {
      int   row  = rowbase + j + 8 * half;
      float hold = h_l[(size_t)row * Dd + col];
      float r = sigf(accR[nt][j] + bR);
      float z = sigf(accZ[nt][j] + bZ);
      float nn = tanhf(accN[nt][j] + bIN + r * (accHN[nt][j] + bHN));
      h_l[(size_t)row * Dd + col] = nn + z * (hold - nn);
    }
  }
}

// ---------------------------------------------------------------------------
// Final: out-LN of h[L-1] at last step, D->H head, residual with data[:, -1].
// ---------------------------------------------------------------------------
__global__ void k_final(const float* __restrict__ hbuf, const float* __restrict__ g,
                        const float* __restrict__ be, const float* __restrict__ W_mlp,
                        const float* __restrict__ b_mlp, const float* __restrict__ data,
                        float* __restrict__ out) {
  int row = blockIdx.x;
  int d   = threadIdx.x;
  int b   = row >> 11;
  int n   = row & (Nn - 1);
  const float* h1 = hbuf + (size_t)BN * Dd;   // layer L-1
  float x = h1[(size_t)row * Dd + d];
  __shared__ float s1[64], s2[64], xs[64];
  s1[d] = x; s2[d] = x * x;
  __syncthreads();
  for (int off = 32; off > 0; off >>= 1) {
    if (d < off) { s1[d] += s1[d + off]; s2[d] += s2[d + off]; }
    __syncthreads();
  }
  float mean = s1[0] * (1.f / Dd);
  float var  = s2[0] * (1.f / Dd) - mean * mean;
  float inv  = rsqrtf(var + 1e-5f);
  xs[d] = (x - mean) * inv * g[d] + be[d];
  __syncthreads();
  if (d < Hh) {
    float acc = b_mlp[d];
    #pragma unroll 8
    for (int k = 0; k < Dd; ++k) acc += xs[k] * W_mlp[k * Hh + d];
    out[((size_t)b * Hh + d) * Nn + n] =
        data[(size_t)(b * Tt + (Tt - 1)) * Nn + n] + acc;
  }
}

// ---------------------------------------------------------------------------
// host driver (graph-capture safe: launches only, all on `stream`)
// ---------------------------------------------------------------------------
extern "C" void kernel_launch(void* const* d_in, const int* in_sizes, int n_in,
                              void* d_out, int out_size, void* d_ws, size_t ws_size,
                              hipStream_t stream) {
  const float* data    = (const float*)d_in[0];
  const int*   timei   = (const int*)  d_in[1];
  const int*   wdayi   = (const int*)  d_in[2];
  const float* W_data  = (const float*)d_in[3];
  const float* b_data  = (const float*)d_in[4];
  const float* emb_t   = (const float*)d_in[5];
  const float* emb_w   = (const float*)d_in[6];
  const float* Amat    = (const float*)d_in[7];
  const float* Wi      = (const float*)d_in[8];
  const float* bi      = (const float*)d_in[9];
  const float* Wh      = (const float*)d_in[10];
  const float* bh      = (const float*)d_in[11];
  const float* ln_g    = (const float*)d_in[12];
  const float* ln_b    = (const float*)d_in[13];
  const float* oln_g   = (const float*)d_in[14];
  const float* oln_b   = (const float*)d_in[15];
  const float* W_mlp   = (const float*)d_in[16];
  const float* b_mlp   = (const float*)d_in[17];

  // workspace carve-up (~75.6 MB total, 256B aligned slabs)
  char*  ws  = (char*)d_ws;
  size_t off = 0;
  auto carve = [&](size_t bytes) -> void* {
    void* p = ws + off;
    off += (bytes + 255) & ~(size_t)255;
    return p;
  };
  bf16*  A_bf = (bf16*) carve((size_t)Nn * Nn * sizeof(bf16));        //  8 MB
  bf16*  WiT  = (bf16*) carve((size_t)Ll * Dd * 3 * Dd * sizeof(bf16));
  bf16*  WhT  = (bf16*) carve((size_t)Ll * Dd * 3 * Dd * sizeof(bf16));
  float* hbuf = (float*)carve((size_t)Ll * BN * Dd * sizeof(float));  // 32 MB
  bf16*  Xcat = (bf16*) carve((size_t)G2 * Nn * Dd * sizeof(bf16));   // 16 MB
  bf16*  Ybf  = (bf16*) carve((size_t)G2 * Nn * Dd * sizeof(bf16));   // 16 MB
  (void)ws_size; (void)in_sizes; (void)n_in; (void)out_size;

  // one-shot converts + hidden-state zero (re-done each call: deterministic)
  k_f2bf <<<(Nn * Nn + 255) / 256, 256, 0, stream>>>(Amat, A_bf, Nn * Nn);
  k_convW<<<(Ll * Dd * 3 * Dd + 255) / 256, 256, 0, stream>>>(Wi, WiT);
  k_convW<<<(Ll * Dd * 3 * Dd + 255) / 256, 256, 0, stream>>>(Wh, WhT);
  k_zero <<<(Ll * BN * Dd + 255) / 256, 256, 0, stream>>>(hbuf, Ll * BN * Dd);

  for (int t = 0; t < Tt; ++t) {
    for (int l = 0; l < Ll; ++l) {
      k_ln<<<BN, 64, 0, stream>>>(data, timei, wdayi, W_data, b_data, emb_t,
                                  emb_w, ln_g, ln_b,
                                  hbuf /* h[0] = layer-1 input */,
                                  hbuf + (size_t)l * BN * Dd, Xcat, l, t);
      k_gconv<<<dim3(Nn / 256, G2), 256, 0, stream>>>(A_bf, Xcat, Ybf);
      k_gate<<<BN / 128, 256, 0, stream>>>(Ybf, WiT, WhT, bi, bh, hbuf, l);
    }
  }
  k_final<<<BN, 64, 0, stream>>>(hbuf, oln_g, oln_b, W_mlp, b_mlp, data,
                                 (float*)d_out);
}
